// AxialTransformerBlock_60481729462546
// MI455X (gfx1250) — matrix operand order repaired and checked by reading, not compiled, not run on hardware
//
#include <hip/hip_runtime.h>

// ---------------------------------------------------------------------------
// Types / helpers
// ---------------------------------------------------------------------------
typedef __attribute__((ext_vector_type(16))) __bf16 bf16x16;
typedef __attribute__((ext_vector_type(8)))  float  f32x8;

union BFrag {
    bf16x16 v;
    uint4   q[2];
    unsigned short u[16];
};

__device__ __forceinline__ unsigned short f2bf(float f) {
    unsigned u = __float_as_uint(f);
    unsigned r = u + 0x7FFFu + ((u >> 16) & 1u);   // round-to-nearest-even
    return (unsigned short)(r >> 16);
}
__device__ __forceinline__ float bf2f(unsigned short h) {
    return __uint_as_float(((unsigned)h) << 16);
}

// A-matrix (16x32 bf16, M x K) fragment load per the ISA 16-bit A layout:
// lanes 0-15: elems 0..7 = K 0..7,  elems 8..15 = K 16..23
// lanes16-31: elems 0..7 = K 8..15, elems 8..15 = K 24..31
__device__ __forceinline__ bf16x16 load_afrag(const unsigned short* row_ptr,
                                              int kbase, int half) {
    BFrag f;
    f.q[0] = *(const uint4*)(row_ptr + kbase + half * 8);
    f.q[1] = *(const uint4*)(row_ptr + kbase + 16 + half * 8);
    return f.v;
}

// B-matrix (32x16 bf16, K x N) fragment: lane holds column n = lane&15,
// elems 0..7 = K 8*half..8*half+7, elems 8..15 = K 16+8*half..16+8*half+7.
__device__ __forceinline__ bf16x16 load_bfrag(const unsigned short* col_ptr,
                                              int kbase, int half) {
    BFrag f;
    f.q[0] = *(const uint4*)(col_ptr + kbase + half * 8);
    f.q[1] = *(const uint4*)(col_ptr + kbase + 16 + half * 8);
    return f.v;
}

__device__ __forceinline__ f32x8 wmma_bf16(bf16x16 a, bf16x16 b, f32x8 c) {
    return __builtin_amdgcn_wmma_f32_16x16x32_bf16(false, a, false, b,
                                                   (short)0, c, false, false);
}

__device__ __forceinline__ void ld8bf(const unsigned short* p, float* d) {
    uint4 u = *(const uint4*)p;
    d[0] = bf2f((unsigned short)(u.x & 0xffff)); d[1] = bf2f((unsigned short)(u.x >> 16));
    d[2] = bf2f((unsigned short)(u.y & 0xffff)); d[3] = bf2f((unsigned short)(u.y >> 16));
    d[4] = bf2f((unsigned short)(u.z & 0xffff)); d[5] = bf2f((unsigned short)(u.z >> 16));
    d[6] = bf2f((unsigned short)(u.w & 0xffff)); d[7] = bf2f((unsigned short)(u.w >> 16));
}

// ---------------------------------------------------------------------------
// fp32 -> bf16 conversion (weights)
// ---------------------------------------------------------------------------
__global__ void f2bf_kernel(const float* __restrict__ src,
                            unsigned short* __restrict__ dst, int n) {
    int i = blockIdx.x * 256 + threadIdx.x;
    if (i < n) dst[i] = f2bf(src[i]);
}

// ---------------------------------------------------------------------------
// LayerNorm over last dim (1024), one wave per row, emits bf16
// ---------------------------------------------------------------------------
__global__ void ln_kernel(const float* __restrict__ x,
                          const float* __restrict__ g,
                          const float* __restrict__ b,
                          unsigned short* __restrict__ out, int rows) {
    const int D = 1024;
    int wave = threadIdx.x >> 5, lane = threadIdx.x & 31;
    int row = blockIdx.x * 8 + wave;
    if (row >= rows) return;
    const float* xr = x + (long)row * D;
    float v[32];
    float s = 0.f;
#pragma unroll
    for (int i = 0; i < 32; ++i) { v[i] = xr[lane + i * 32]; s += v[i]; }
#pragma unroll
    for (int m = 1; m < 32; m <<= 1) s += __shfl_xor(s, m);
    float mu = s * (1.0f / 1024.0f);
    float vs = 0.f;
#pragma unroll
    for (int i = 0; i < 32; ++i) { float d = v[i] - mu; vs += d * d; }
#pragma unroll
    for (int m = 1; m < 32; m <<= 1) vs += __shfl_xor(vs, m);
    float rs = rsqrtf(vs * (1.0f / 1024.0f) + 1e-5f);
    unsigned short* orow = out + (long)row * D;
#pragma unroll
    for (int i = 0; i < 32; ++i) {
        int d = lane + i * 32;
        orow[d] = f2bf((v[i] - mu) * rs * g[d] + b[d]);
    }
}

// ---------------------------------------------------------------------------
// Generic bf16 WMMA GEMM:  OUT[M,N] = A[M,K] * W[N,K]^T  (+ epilogue)
// Wave tile = 32 rows x 64 cols. perm: 0 row-major, 1 -> [c,h,s,d] (Q/K attn
// layout), 2 -> [c,h,d,s] (V transposed layout).
// ---------------------------------------------------------------------------
#define GF_BIAS 1
#define GF_RELU 2
#define GF_RES  4
#define GF_OUTF 8
#define GF_OUTB 16

__global__ void gemm_bf16_kernel(const unsigned short* __restrict__ A,
                                 const unsigned short* __restrict__ W,
                                 const float* __restrict__ bias,
                                 const float* __restrict__ res,
                                 float* __restrict__ outF,
                                 unsigned short* __restrict__ outB,
                                 int M, int N, int K, int flags, int perm) {
    int wave = threadIdx.x >> 5, lane = threadIdx.x & 31;
    int half = lane >> 4, ln = lane & 15;
    int nt = N >> 6;
    int g  = blockIdx.x * 8 + wave;
    int m0 = (g / nt) * 32;
    int n0 = (g % nt) * 64;
    if (m0 >= M) return;

    const unsigned short* arow0 = A + (long)(m0 + ln) * K;
    const unsigned short* arow1 = A + (long)(m0 + 16 + ln) * K;
    const unsigned short* brow0 = W + (long)(n0 + 0 * 16 + ln) * K;
    const unsigned short* brow1 = W + (long)(n0 + 1 * 16 + ln) * K;
    const unsigned short* brow2 = W + (long)(n0 + 2 * 16 + ln) * K;
    const unsigned short* brow3 = W + (long)(n0 + 3 * 16 + ln) * K;

    f32x8 acc[2][4];
#pragma unroll
    for (int i = 0; i < 2; ++i)
#pragma unroll
        for (int t = 0; t < 4; ++t)
            acc[i][t] = (f32x8){0.f, 0.f, 0.f, 0.f, 0.f, 0.f, 0.f, 0.f};

    for (int k = 0; k < K; k += 32) {
        bf16x16 a0 = load_afrag(arow0, k, half);
        bf16x16 a1 = load_afrag(arow1, k, half);
        bf16x16 b0 = load_bfrag(brow0, k, half);
        bf16x16 b1 = load_bfrag(brow1, k, half);
        bf16x16 b2 = load_bfrag(brow2, k, half);
        bf16x16 b3 = load_bfrag(brow3, k, half);
        acc[0][0] = wmma_bf16(a0, b0, acc[0][0]);
        acc[1][0] = wmma_bf16(a1, b0, acc[1][0]);
        acc[0][1] = wmma_bf16(a0, b1, acc[0][1]);
        acc[1][1] = wmma_bf16(a1, b1, acc[1][1]);
        acc[0][2] = wmma_bf16(a0, b2, acc[0][2]);
        acc[1][2] = wmma_bf16(a1, b2, acc[1][2]);
        acc[0][3] = wmma_bf16(a0, b3, acc[0][3]);
        acc[1][3] = wmma_bf16(a1, b3, acc[1][3]);
    }

#pragma unroll
    for (int i = 0; i < 2; ++i)
#pragma unroll
        for (int t = 0; t < 4; ++t)
#pragma unroll
            for (int r = 0; r < 8; ++r) {
                int row = m0 + i * 16 + r + half * 8;
                int col = n0 + t * 16 + ln;
                float v = acc[i][t][r];
                if (flags & GF_BIAS) v += bias[col];
                if (flags & GF_RELU) v = fmaxf(v, 0.f);
                if (flags & GF_RES)  v += res[(long)row * N + col];
                long idx;
                if (perm == 0) {
                    idx = (long)row * N + col;
                } else {
                    int s = row >> 2, c = row & 3;
                    int h = col >> 6, dd = col & 63;
                    if (perm == 1)
                        idx = (((long)(c * 16 + h)) * 4096 + s) * 64 + dd;
                    else
                        idx = (((long)(c * 16 + h)) * 64 + dd) * 4096 + s;
                }
                if (flags & GF_OUTF) outF[idx] = v;
                if (flags & GF_OUTB) outB[idx] = f2bf(v);
            }
}

// ---------------------------------------------------------------------------
// Channel attention: seq length C=4, H_C=4 heads, hd=256. One wave per (s,h).
// ---------------------------------------------------------------------------
__global__ void chanattn_kernel(const unsigned short* __restrict__ Q,
                                const unsigned short* __restrict__ K,
                                const unsigned short* __restrict__ V,
                                unsigned short* __restrict__ O) {
    int wave = threadIdx.x >> 5, lane = threadIdx.x & 31;
    int g = blockIdx.x * 8 + wave;           // 0 .. 16383 = s*4 + h
    int s = g >> 2, h = g & 3;
    long base = ((long)s * 4) * 1024 + h * 256 + lane * 8;

    float q[4][8], k[4][8], v[4][8];
#pragma unroll
    for (int c = 0; c < 4; ++c) {
        ld8bf(Q + base + c * 1024, q[c]);
        ld8bf(K + base + c * 1024, k[c]);
        ld8bf(V + base + c * 1024, v[c]);
    }
    float sc[4][4];
#pragma unroll
    for (int i = 0; i < 4; ++i)
#pragma unroll
        for (int j = 0; j < 4; ++j) {
            float a = 0.f;
#pragma unroll
            for (int d = 0; d < 8; ++d) a += q[i][d] * k[j][d];
#pragma unroll
            for (int m = 1; m < 32; m <<= 1) a += __shfl_xor(a, m);
            sc[i][j] = a * 0.0625f;          // 1/sqrt(256)
        }
#pragma unroll
    for (int i = 0; i < 4; ++i) {
        float mx = fmaxf(fmaxf(sc[i][0], sc[i][1]), fmaxf(sc[i][2], sc[i][3]));
        float sum = 0.f;
#pragma unroll
        for (int j = 0; j < 4; ++j) { sc[i][j] = __expf(sc[i][j] - mx); sum += sc[i][j]; }
        float inv = 1.f / sum;
#pragma unroll
        for (int j = 0; j < 4; ++j) sc[i][j] *= inv;
    }
#pragma unroll
    for (int i = 0; i < 4; ++i) {
        unsigned short* op = O + base + i * 1024;
#pragma unroll
        for (int d = 0; d < 8; ++d) {
            float o = sc[i][0] * v[0][d] + sc[i][1] * v[1][d] +
                      sc[i][2] * v[2][d] + sc[i][3] * v[3][d];
            op[d] = f2bf(o);
        }
    }
}

// ---------------------------------------------------------------------------
// RoPE in-place on Q/K laid out as [c*16+h][s][64] bf16 (pairs 2j, 2j+1)
// ---------------------------------------------------------------------------
__global__ void rope_kernel(unsigned short* __restrict__ Qb,
                            unsigned short* __restrict__ Kb,
                            const int* __restrict__ positions) {
    long idx = (long)blockIdx.x * 256 + threadIdx.x;  // 64*4096*32 threads
    int j = (int)(idx & 31);
    long r = idx >> 5;
    int s = (int)(r & 4095);
    int ch = (int)(r >> 12);
    float pos = (float)positions[s];
    float inv = __powf(10000.0f, -(float)j / 32.0f);
    float ang = pos * inv;
    float ca = __cosf(ang), sa = __sinf(ang);
    long base = (((long)ch * 4096) + s) * 64 + 2 * j;
    float e = bf2f(Qb[base]), o = bf2f(Qb[base + 1]);
    Qb[base]     = f2bf(e * ca - o * sa);
    Qb[base + 1] = f2bf(e * sa + o * ca);
    e = bf2f(Kb[base]); o = bf2f(Kb[base + 1]);
    Kb[base]     = f2bf(e * ca - o * sa);
    Kb[base + 1] = f2bf(e * sa + o * ca);
}

// ---------------------------------------------------------------------------
// Causal flash attention, temporal axis. One wave = 16 queries of one (c,h).
// Q/K in [ch][s][64] bf16; V transposed [ch][64][s] bf16.
// Output bf16 in (s, c, d=h*64+dim) row-major for the O projection.
// ---------------------------------------------------------------------------
__global__ void flashattn_kernel(const unsigned short* __restrict__ Qb,
                                 const unsigned short* __restrict__ Kb,
                                 const unsigned short* __restrict__ Vt,
                                 unsigned short* __restrict__ Out) {
    __shared__ unsigned short psh[8][16 * 40];   // per-wave 16x32 bf16 (stride 40)
    int wave = threadIdx.x >> 5, lane = threadIdx.x & 31;
    int half = lane >> 4, ln = lane & 15;
    int ch = blockIdx.x >> 5;                    // 64 (c,h) pairs
    int qblk = blockIdx.x & 31;                  // 32 blocks of 128 queries
    int q0 = qblk * 128 + wave * 16;

    const unsigned short* Qch = Qb + (long)ch * 4096 * 64;
    const unsigned short* Kch = Kb + (long)ch * 4096 * 64;
    const unsigned short* Vch = Vt + (long)ch * 64 * 4096;
    unsigned short* myp = &psh[wave][0];

    const unsigned short* qrow = Qch + (long)(q0 + ln) * 64;
    bf16x16 aq0 = load_afrag(qrow, 0, half);
    bf16x16 aq1 = load_afrag(qrow, 32, half);

    f32x8 acc[4];
#pragma unroll
    for (int t = 0; t < 4; ++t) acc[t] = (f32x8){0.f,0.f,0.f,0.f,0.f,0.f,0.f,0.f};
    float mst[8], lst[8];
#pragma unroll
    for (int r = 0; r < 8; ++r) { mst[r] = -1e30f; lst[r] = 0.f; }

    for (int kb = 0; kb < q0 + 16; kb += 32) {
        float sc[2][8];
#pragma unroll
        for (int j = 0; j < 2; ++j) {
            const unsigned short* krow = Kch + (long)(kb + j * 16 + ln) * 64;
            bf16x16 b0 = load_bfrag(krow, 0, half);
            bf16x16 b1 = load_bfrag(krow, 32, half);
            f32x8 z = (f32x8){0.f,0.f,0.f,0.f,0.f,0.f,0.f,0.f};
            z = wmma_bf16(aq0, b0, z);
            z = wmma_bf16(aq1, b1, z);
            int ck = kb + j * 16 + ln;           // key position (this lane's col)
#pragma unroll
            for (int r = 0; r < 8; ++r) {
                int rq = q0 + r + half * 8;      // query position (this row)
                float v = z[r] * 0.125f;         // 1/sqrt(64)
                sc[j][r] = (ck > rq) ? -1e30f : v;
            }
        }
        float rmax[8], rsum[8], alpha[8];
#pragma unroll
        for (int r = 0; r < 8; ++r) {
            float mr = fmaxf(sc[0][r], sc[1][r]);
#pragma unroll
            for (int m = 1; m < 16; m <<= 1) mr = fmaxf(mr, __shfl_xor(mr, m));
            rmax[r] = mr;
            float mn = fmaxf(mst[r], mr);
            alpha[r] = __expf(mst[r] - mn);
            mst[r] = mn;
        }
#pragma unroll
        for (int r = 0; r < 8; ++r) {
            sc[0][r] = __expf(sc[0][r] - mst[r]);
            sc[1][r] = __expf(sc[1][r] - mst[r]);
            float ps = sc[0][r] + sc[1][r];
#pragma unroll
            for (int m = 1; m < 16; m <<= 1) ps += __shfl_xor(ps, m);
            rsum[r] = ps;
            lst[r] = lst[r] * alpha[r] + rsum[r];
        }
#pragma unroll
        for (int t = 0; t < 4; ++t)
#pragma unroll
            for (int r = 0; r < 8; ++r) acc[t][r] *= alpha[r];

        // route P through LDS: C-layout (f32) -> A-layout (bf16)
#pragma unroll
        for (int j = 0; j < 2; ++j)
#pragma unroll
            for (int r = 0; r < 8; ++r)
                myp[(r + half * 8) * 40 + j * 16 + ln] = f2bf(sc[j][r]);
        __builtin_amdgcn_wave_barrier();
        BFrag pf;
#pragma unroll
        for (int e = 0; e < 16; ++e) {
            int kk = (e < 8) ? (half * 8 + e) : (16 + half * 8 + (e - 8));
            pf.u[e] = myp[ln * 40 + kk];
        }
        __builtin_amdgcn_wave_barrier();

        // P(16x32) * V(32x64): Vt gives contiguous K per output dim
#pragma unroll
        for (int t = 0; t < 4; ++t) {
            const unsigned short* vrow = Vch + (long)(t * 16 + ln) * 4096;
            bf16x16 bv = load_bfrag(vrow, kb, half);
            acc[t] = wmma_bf16(pf.v, bv, acc[t]);
        }
    }

    int c = ch >> 4, h = ch & 15;
#pragma unroll
    for (int r = 0; r < 8; ++r) {
        float inv = 1.f / lst[r];
        int srow = q0 + r + half * 8;
#pragma unroll
        for (int t = 0; t < 4; ++t) {
            long o = ((long)srow * 4 + c) * 1024 + h * 64 + t * 16 + ln;
            Out[o] = f2bf(acc[t][r] * inv);
        }
    }
}

// ---------------------------------------------------------------------------
// Host orchestration
// ---------------------------------------------------------------------------
extern "C" void kernel_launch(void* const* d_in, const int* in_sizes, int n_in,
                              void* d_out, int out_size, void* d_ws, size_t ws_size,
                              hipStream_t stream) {
    (void)in_sizes; (void)n_in; (void)out_size; (void)ws_size;
    const int M = 16384;          // S*C rows
    const int D = 1024;

    const float* x_in      = (const float*)d_in[0];
    const int*   positions = (const int*)d_in[1];
    const float* g_c = (const float*)d_in[10];
    const float* b_c = (const float*)d_in[11];
    const float* g_t = (const float*)d_in[12];
    const float* b_t = (const float*)d_in[13];
    const float* g_m = (const float*)d_in[14];
    const float* b_m = (const float*)d_in[15];
    const float* b1  = (const float*)d_in[17];
    const float* b2  = (const float*)d_in[19];

    char* ws = (char*)d_ws;
    size_t off = 0;
    auto alloc = [&](size_t bytes) {
        void* p = ws + off;
        off += (bytes + 255) & ~(size_t)255;
        return p;
    };
    // bf16 weights
    unsigned short* w_qc = (unsigned short*)alloc((size_t)D * D * 2);
    unsigned short* w_kc = (unsigned short*)alloc((size_t)D * D * 2);
    unsigned short* w_vc = (unsigned short*)alloc((size_t)D * D * 2);
    unsigned short* w_oc = (unsigned short*)alloc((size_t)D * D * 2);
    unsigned short* w_qt = (unsigned short*)alloc((size_t)D * D * 2);
    unsigned short* w_kt = (unsigned short*)alloc((size_t)D * D * 2);
    unsigned short* w_vt = (unsigned short*)alloc((size_t)D * D * 2);
    unsigned short* w_ot = (unsigned short*)alloc((size_t)D * D * 2);
    unsigned short* w_1  = (unsigned short*)alloc((size_t)4 * D * D * 2);
    unsigned short* w_2  = (unsigned short*)alloc((size_t)4 * D * D * 2);
    // fp32 residual stream
    float* xbuf = (float*)alloc((size_t)M * D * 4);
    // bf16 staging
    unsigned short* B1 = (unsigned short*)alloc((size_t)M * D * 2);
    unsigned short* B2 = (unsigned short*)alloc((size_t)M * D * 2);
    unsigned short* B3 = (unsigned short*)alloc((size_t)M * D * 2);
    unsigned short* B4 = (unsigned short*)alloc((size_t)M * D * 2);
    unsigned short* H  = (unsigned short*)alloc((size_t)M * 4 * D * 2);   // MLP hidden

    // 1) convert weights to bf16
    {
        const int widx[10]  = {2, 3, 4, 5, 6, 7, 8, 9, 16, 18};
        unsigned short* wdst[10] = {w_qc, w_kc, w_vc, w_oc, w_qt, w_kt, w_vt, w_ot, w_1, w_2};
        for (int i = 0; i < 10; ++i) {
            int n = (i < 8) ? D * D : 4 * D * D;
            f2bf_kernel<<<(n + 255) / 256, 256, 0, stream>>>(
                (const float*)d_in[widx[i]], wdst[i], n);
        }
    }

    // 2) LN1 -> B1, channel QKV projections
    ln_kernel<<<M / 8, 256, 0, stream>>>(x_in, g_c, b_c, B1, M);
    gemm_bf16_kernel<<<1024, 256, 0, stream>>>(B1, w_qc, nullptr, nullptr, nullptr, B2,
                                               M, D, D, GF_OUTB, 0);
    gemm_bf16_kernel<<<1024, 256, 0, stream>>>(B1, w_kc, nullptr, nullptr, nullptr, B3,
                                               M, D, D, GF_OUTB, 0);
    gemm_bf16_kernel<<<1024, 256, 0, stream>>>(B1, w_vc, nullptr, nullptr, nullptr, B4,
                                               M, D, D, GF_OUTB, 0);
    // 3) channel attention -> B1, then O-projection with residual (x_in) -> xbuf
    chanattn_kernel<<<M / 8, 256, 0, stream>>>(B2, B3, B4, B1);
    gemm_bf16_kernel<<<1024, 256, 0, stream>>>(B1, w_oc, nullptr, x_in, xbuf, nullptr,
                                               M, D, D, GF_RES | GF_OUTF, 0);

    // 4) LN2 -> B2, temporal QKV projections with permuted stores
    ln_kernel<<<M / 8, 256, 0, stream>>>(xbuf, g_t, b_t, B2, M);
    gemm_bf16_kernel<<<1024, 256, 0, stream>>>(B2, w_qt, nullptr, nullptr, nullptr, B1,
                                               M, D, D, GF_OUTB, 1);   // Q [ch][s][d]
    gemm_bf16_kernel<<<1024, 256, 0, stream>>>(B2, w_kt, nullptr, nullptr, nullptr, B3,
                                               M, D, D, GF_OUTB, 1);   // K [ch][s][d]
    gemm_bf16_kernel<<<1024, 256, 0, stream>>>(B2, w_vt, nullptr, nullptr, nullptr, B4,
                                               M, D, D, GF_OUTB, 2);   // V [ch][d][s]
    // 5) RoPE in-place on Q/K
    rope_kernel<<<(64 * 4096 * 32) / 256, 256, 0, stream>>>(B1, B3, positions);
    // 6) causal flash attention -> B2, O-projection with residual (in-place xbuf)
    flashattn_kernel<<<64 * 32, 256, 0, stream>>>(B1, B3, B4, B2);
    gemm_bf16_kernel<<<1024, 256, 0, stream>>>(B2, w_ot, nullptr, xbuf, xbuf, nullptr,
                                               M, D, D, GF_RES | GF_OUTF, 0);

    // 7) LN3 -> B1, MLP
    ln_kernel<<<M / 8, 256, 0, stream>>>(xbuf, g_m, b_m, B1, M);
    gemm_bf16_kernel<<<4096, 256, 0, stream>>>(B1, w_1, b1, nullptr, nullptr, H,
                                               M, 4 * D, D, GF_BIAS | GF_RELU | GF_OUTB, 0);
    gemm_bf16_kernel<<<1024, 256, 0, stream>>>(H, w_2, b2, xbuf, (float*)d_out, nullptr,
                                               M, D, 4 * D, GF_BIAS | GF_RES | GF_OUTF, 0);
}
